// BottleneckBlock_58213986730228
// MI455X (gfx1250) — compile-verified, matrix operand order checked by reading, not run on hardware
//
#include <hip/hip_runtime.h>
#include <hip/hip_bf16.h>
#include <stdint.h>

// ---------------- problem constants (from reference) ----------------
constexpr int   N_  = 8;
constexpr int   D_  = 64;
constexpr int   T_  = 8192;
constexpr int   K_  = 2048;
constexpr int   NT_ = N_ * T_;          // 65536
constexpr float MU_     = 0.99f;
constexpr float THRESH_ = 1.0f;

// d_out layout (floats), reference return order:
// (xd, codes, commit, k_new, k_sum_n, k_elem_n)
constexpr size_t OFF_XD     = 0;
constexpr size_t OFF_CODES  = OFF_XD    + (size_t)N_ * D_ * T_;   // 4194304
constexpr size_t OFF_COMMIT = OFF_CODES + (size_t)NT_;            // 4259840
constexpr size_t OFF_KNEW   = OFF_COMMIT + 1;                     // 4259841
constexpr size_t OFF_KSUM   = OFF_KNEW  + (size_t)K_ * D_;        // 4390913
constexpr size_t OFF_KELEM  = OFF_KSUM  + (size_t)K_ * D_;        // 4521985

// workspace layout (bytes)
constexpr size_t WS_KH     = 0;                         // K*D f16 = 262144 B
constexpr size_t WS_HN     = (size_t)K_ * D_ * 2;       // K floats = 8192 B
constexpr size_t WS_COMMIT = WS_HN + (size_t)K_ * 4;    // 1 float

typedef _Float16 v16h __attribute__((ext_vector_type(16)));
typedef _Float16 v8h  __attribute__((ext_vector_type(8)));
typedef float    v8f  __attribute__((ext_vector_type(8)));
typedef int      v4i_ __attribute__((vector_size(4 * sizeof(int))));

// --------- CDNA5 async Global->LDS path (guarded: falls back to sync) ----
#if defined(__has_builtin)
#if __has_builtin(__builtin_amdgcn_global_load_async_to_lds_b128) && \
    __has_builtin(__builtin_amdgcn_s_wait_asynccnt)
#define VQ_ASYNC 1
#endif
#endif
#ifndef VQ_ASYNC
#define VQ_ASYNC 0
#endif

typedef __attribute__((address_space(1))) v4i_* gv4i_p;   // global AS
typedef __attribute__((address_space(3))) v4i_* lv4i_p;   // LDS AS

__device__ __forceinline__ void vq_wait_stage() {
#if VQ_ASYNC
    __builtin_amdgcn_s_wait_asynccnt(0);
#endif
}

// ---------------- kernel 1: f16 codebook + half norms ----------------
// One wave per codebook row: lane-parallel over D (coalesced), shfl reduce.
__global__ __launch_bounds__(256)
void vq_prep_kernel(const float* __restrict__ kcb,
                    _Float16* __restrict__ kh,
                    float* __restrict__ hn) {
    const int wid  = (blockIdx.x * 256 + threadIdx.x) >> 5;   // row
    const int lane = threadIdx.x & 31;
    const float* row = kcb + (size_t)wid * D_;
    float v0 = row[lane];
    float v1 = row[lane + 32];
    kh[(size_t)wid * D_ + lane]      = (_Float16)v0;
    kh[(size_t)wid * D_ + lane + 32] = (_Float16)v1;
    float s = v0 * v0 + v1 * v1;
    #pragma unroll
    for (int off = 16; off >= 1; off >>= 1) s += __shfl_xor(s, off, 32);
    if (lane == 0) hn[wid] = 0.5f * s;
}

// ---------------- kernel 2: fused distance GEMM + argmin (WMMA) ------
// WG = 256 threads (8 waves); WG covers 128 consecutive t for one n.
// Codebook staged in 128-entry chunks, double-buffered with async loads.
constexpr int WG_ROWS   = 128;
constexpr int XS_STRIDE = 72;   // f16 units; keeps 16B alignment, skews banks
constexpr int CHUNK     = 128;  // codebook entries per staged chunk
constexpr int NCHUNK    = K_ / CHUNK;   // 16

__device__ __forceinline__
void vq_stage_chunk(const _Float16* __restrict__ kh,
                    const float* __restrict__ hn,
                    _Float16* ksbuf, float* hnsbuf, int cb, int tid) {
    #pragma unroll
    for (int it = 0; it < (CHUNK * D_ * 2) / (256 * 16); ++it) {  // 4 x 16B
        int u = it * 256 + tid;
        int r = u >> 3, c2 = u & 7;
        const _Float16* src = &kh[(size_t)(cb + r) * D_ + c2 * 8];
        _Float16*       dst = &ksbuf[r * XS_STRIDE + c2 * 8];
#if VQ_ASYNC
        __builtin_amdgcn_global_load_async_to_lds_b128(
            (gv4i_p)(uintptr_t)src,
            (lv4i_p)(uint32_t)(uintptr_t)dst,
            0, 0);
#else
        *(v8h*)dst = *(const v8h*)src;
#endif
    }
    if (tid < CHUNK) hnsbuf[tid] = hn[cb + tid];
}

__global__ __launch_bounds__(256)
void vq_argmin_kernel(const float* __restrict__ x,
                      const _Float16* __restrict__ kh,
                      const float* __restrict__ hn,
                      float* __restrict__ codes_out) {
    __shared__ _Float16 xs[WG_ROWS * XS_STRIDE];
    __shared__ _Float16 ks[2][CHUNK * XS_STRIDE];
    __shared__ float    hns[2][CHUNK];

    const int tid     = threadIdx.x;
    const int tchunks = T_ / WG_ROWS;                 // 64
    const int n       = blockIdx.x / tchunks;
    const int t0      = (blockIdx.x % tchunks) * WG_ROWS;

    // Stage x tile: [128 t][64 d] row-major f16 (global reads coalesced in t).
    const float* xb = x + (size_t)n * D_ * T_ + t0;
    #pragma unroll
    for (int it = 0; it < (WG_ROWS * D_) / 256; ++it) {   // 32 iters
        int idx = it * 256 + tid;
        int d   = idx >> 7;
        int tl  = idx & (WG_ROWS - 1);
        xs[tl * XS_STRIDE + d] = (_Float16)xb[(size_t)d * T_ + tl];
    }

    // Kick off chunk 0 into buffer 0, then one wait+barrier for everything.
    vq_stage_chunk(kh, hn, ks[0], hns[0], 0, tid);
    vq_wait_stage();
    __syncthreads();

    const int wave = tid >> 5;
    const int lane = tid & 31;
    const int ln16 = lane & 15;
    const int hi   = lane >> 4;
    const int d0a  = hi ? 8 : 0;   // 16-bit A layout: lanes16-31 hold K=8..15/24..31

    // A operand for this wave's 16 rows, dims 0..31 (a0) and 32..63 (a1).
    const _Float16* arow = &xs[(wave * 16 + ln16) * XS_STRIDE];
    v16h a0, a1;
    ((v8h*)&a0)[0] = *(const v8h*)(arow + d0a);
    ((v8h*)&a0)[1] = *(const v8h*)(arow + d0a + 16);
    ((v8h*)&a1)[0] = *(const v8h*)(arow + 32 + d0a);
    ((v8h*)&a1)[1] = *(const v8h*)(arow + 32 + d0a + 16);

    float minv[8];
    int   mini[8];
    #pragma unroll
    for (int r = 0; r < 8; ++r) { minv[r] = 3.4e38f; mini[r] = 0; }

    for (int c = 0; c < NCHUNK; ++c) {
        const int cb = c * CHUNK;

        // Overlap: fetch next chunk into the other buffer while we compute.
        if (c + 1 < NCHUNK)
            vq_stage_chunk(kh, hn, ks[(c + 1) & 1], hns[(c + 1) & 1],
                           cb + CHUNK, tid);
        if (c + 2 < NCHUNK)   // speculative L2 prefetch two chunks ahead
            __builtin_prefetch(&kh[(size_t)(cb + 2 * CHUNK) * D_], 0, 1);

        const _Float16* ksb = ks[c & 1];
        const float*    hnb = hns[c & 1];

        #pragma unroll
        for (int sub = 0; sub < CHUNK / 16; ++sub) {
            // B tile loaded with the same per-lane pattern as A => B = tile^T,
            // so C[m][n] = dot(x_m, k_n).
            const _Float16* brow = &ksb[(sub * 16 + ln16) * XS_STRIDE];
            v16h b0, b1;
            ((v8h*)&b0)[0] = *(const v8h*)(brow + d0a);
            ((v8h*)&b0)[1] = *(const v8h*)(brow + d0a + 16);
            ((v8h*)&b1)[0] = *(const v8h*)(brow + 32 + d0a);
            ((v8h*)&b1)[1] = *(const v8h*)(brow + 32 + d0a + 16);

            v8f cacc = {};
            cacc = __builtin_amdgcn_wmma_f32_16x16x32_f16(
                       false, a0, false, b0, (short)0, cacc, false, false);
            cacc = __builtin_amdgcn_wmma_f32_16x16x32_f16(
                       false, a1, false, b1, (short)0, cacc, false, false);

            float h    = hnb[sub * 16 + ln16];
            int   cand = cb + sub * 16 + ln16;
            #pragma unroll
            for (int r = 0; r < 8; ++r) {
                float s = h - cacc[r];                 // 0.5||k||^2 - x.k
                if (s < minv[r]) { minv[r] = s; mini[r] = cand; }
            }
        }

        // My async loads for chunk c+1 done; all waves done reading old buffer.
        vq_wait_stage();
        __syncthreads();
    }

    // Cross-lane argmin within each 16-lane half (C layout: lane = column).
    #pragma unroll
    for (int off = 8; off >= 1; off >>= 1) {
        #pragma unroll
        for (int r = 0; r < 8; ++r) {
            float ov = __shfl_xor(minv[r], off, 32);
            int   oi = __shfl_xor(mini[r], off, 32);
            if (ov < minv[r] || (ov == minv[r] && oi < mini[r])) {
                minv[r] = ov; mini[r] = oi;
            }
        }
    }

    // lane 0 holds rows 0..7 (VGPR r -> row r), lane 16 holds rows 8..15.
    if (ln16 == 0) {
        size_t base = (size_t)n * T_ + t0 + wave * 16 + hi * 8;
        #pragma unroll
        for (int r = 0; r < 8; ++r) codes_out[base + r] = (float)mini[r];
    }
}

// ---------------- kernel 3: xd = k[codes] (straight-through) + commit ----
__global__ __launch_bounds__(256)
void vq_gather_kernel(const float* __restrict__ x,
                      const float* __restrict__ kcb,
                      const float* __restrict__ codes_f,
                      float* __restrict__ xd,
                      float* __restrict__ commit_ws) {
    __shared__ float red[256];
    const int tchunks = T_ / 256;                 // 32
    const int n = blockIdx.x / tchunks;
    const int t = (blockIdx.x % tchunks) * 256 + threadIdx.x;
    const int c = (int)codes_f[(size_t)n * T_ + t];

    const float*  xb  = x   + (size_t)n * D_ * T_ + t;
    float*        ob  = xd  + (size_t)n * D_ * T_ + t;
    const float4* kr4 = (const float4*)(kcb + (size_t)c * D_);  // 256B aligned

    float acc = 0.f;
    #pragma unroll
    for (int dq = 0; dq < D_ / 4; ++dq) {
        float4 kv = kr4[dq];
        float kvv[4] = {kv.x, kv.y, kv.z, kv.w};
        #pragma unroll
        for (int q = 0; q < 4; ++q) {
            int d = dq * 4 + q;
            float xv = xb[(size_t)d * T_];
            ob[(size_t)d * T_] = kvv[q];   // forward value of straight-through
            float df = kvv[q] - xv;
            acc += df * df;
        }
    }
    red[threadIdx.x] = acc;
    __syncthreads();
    for (int s = 128; s > 0; s >>= 1) {
        if (threadIdx.x < s) red[threadIdx.x] += red[threadIdx.x + s];
        __syncthreads();
    }
    if (threadIdx.x == 0) atomicAdd(commit_ws, red[0]);
}

// ---------------- kernel 4: scatter (1-MU)*xf into EMA accumulators ------
__global__ __launch_bounds__(256)
void vq_scatter_kernel(const float* __restrict__ x,
                       const float* __restrict__ codes_f,
                       float* __restrict__ ksum_acc,
                       float* __restrict__ kelem_acc) {
    const int tchunks = T_ / 256;
    const int n = blockIdx.x / tchunks;
    const int t = (blockIdx.x % tchunks) * 256 + threadIdx.x;
    const int c = (int)codes_f[(size_t)n * T_ + t];
    const float w = 1.0f - MU_;
    atomicAdd(&kelem_acc[c], w);
    const float* xb = x + (size_t)n * D_ * T_ + t;
    float* sb = ksum_acc + (size_t)c * D_;
    #pragma unroll 8
    for (int d = 0; d < D_; ++d)
        atomicAdd(&sb[d], w * xb[(size_t)d * T_]);
}

// ---------------- kernel 5: EMA finalize (in place, single writer) -------
__global__ void vq_ema_kernel(const float* __restrict__ ksum_old,
                              const float* __restrict__ kelem_old,
                              float* __restrict__ ksum_n,
                              float* __restrict__ kelem_n) {
    int i = blockIdx.x * blockDim.x + threadIdx.x;   // 0..K*D
    ksum_n[i] = MU_ * ksum_old[i] + ksum_n[i];
    if (i < K_) kelem_n[i] = MU_ * kelem_old[i] + kelem_n[i];
}

// ---------------- kernel 6: k_new + commit scalar ------------------------
__global__ void vq_knew_kernel(const float* __restrict__ kcb,
                               const float* __restrict__ ksum_n,
                               const float* __restrict__ kelem_n,
                               float* __restrict__ knew,
                               const float* __restrict__ commit_ws,
                               float* __restrict__ commit_out) {
    int i = blockIdx.x * blockDim.x + threadIdx.x;   // 0..K*D
    int j = i >> 6;                                  // / D
    float e = kelem_n[j];
    knew[i] = (e >= THRESH_) ? (ksum_n[i] / e) : kcb[i];
    if (i == 0) *commit_out = *commit_ws * (1.0f / (float)((size_t)NT_ * D_));
}

// ---------------- launch ------------------------------------------------
extern "C" void kernel_launch(void* const* d_in, const int* in_sizes, int n_in,
                              void* d_out, int out_size, void* d_ws, size_t ws_size,
                              hipStream_t stream) {
    const float* x       = (const float*)d_in[0];   // [N, D, T]
    const float* kcb     = (const float*)d_in[1];   // [K, D]
    const float* ksum_o  = (const float*)d_in[2];   // [K, D]
    const float* kelem_o = (const float*)d_in[3];   // [K]

    float* out = (float*)d_out;
    float* xd      = out + OFF_XD;
    float* codes_f = out + OFF_CODES;
    float* commit  = out + OFF_COMMIT;
    float* knew    = out + OFF_KNEW;
    float* ksum_n  = out + OFF_KSUM;
    float* kelem_n = out + OFF_KELEM;

    uint8_t* ws = (uint8_t*)d_ws;
    _Float16* kh     = (_Float16*)(ws + WS_KH);
    float*    hn     = (float*)   (ws + WS_HN);
    float*    cmt_ws = (float*)   (ws + WS_COMMIT);

    // zero accumulators (graph-capturable memset nodes)
    (void)hipMemsetAsync(ksum_n,  0, (size_t)K_ * D_ * sizeof(float), stream);
    (void)hipMemsetAsync(kelem_n, 0, (size_t)K_ * sizeof(float),      stream);
    (void)hipMemsetAsync(cmt_ws,  0, sizeof(float),                   stream);

    // 1. f16 codebook + half norms (wave per row)
    vq_prep_kernel<<<K_ / 8, 256, 0, stream>>>(kcb, kh, hn);

    // 2. fused distance GEMM + argmin (WMMA + async double-buffered staging)
    vq_argmin_kernel<<<(N_ * T_) / WG_ROWS, 256, 0, stream>>>(x, kh, hn, codes_f);

    // 3. gather xd + commit partial
    vq_gather_kernel<<<(N_ * T_) / 256, 256, 0, stream>>>(x, kcb, codes_f, xd, cmt_ws);

    // 4. scatter EMA accumulators
    vq_scatter_kernel<<<(N_ * T_) / 256, 256, 0, stream>>>(x, codes_f, ksum_n, kelem_n);

    // 5. EMA finalize
    vq_ema_kernel<<<(K_ * D_) / 256, 256, 0, stream>>>(ksum_o, kelem_o, ksum_n, kelem_n);

    // 6. k_new + commit scalar
    vq_knew_kernel<<<(K_ * D_) / 256, 256, 0, stream>>>(kcb, ksum_n, kelem_n, knew,
                                                        cmt_ws, commit);
}
